// MultiHeadAttentionLayer_59485297050177
// MI455X (gfx1250) — matrix operand order repaired and checked by reading, not compiled, not run on hardware
//
#include <hip/hip_runtime.h>

// ---------------- geometry ----------------
#define F_DIM 40
#define E_DIM 64
#define NH    4
#define A_DIM 64
#define STR   72      // bf16-element row stride for Xs/Qs/Ks/attn/VT (bank-friendly, mult of 8)
#define SSTR  50      // f32 stride for scores scratch

// LDS offsets in ushort (bf16) units
#define XS_OFF     0                          // 48 x 72
#define QS_OFF(h)  (3456 + (h)*6912)          // 48 x 72 per head
#define KS_OFF(h)  (QS_OFF(h) + 3456)         // 48 x 72 per head
#define ZB_OFF     31104                      // start of zero-initialized region
#define ATT_OFF(h) (ZB_OFF + (h)*3456)        // 48 x 72 per head (zero padded)
#define VT_OFF(h)  (ZB_OFF + 13824 + (h)*4608)// 64 x 72 per head (zero padded)
#define LDS_USH    63360                      // 126720 bytes total

typedef __bf16 bf16_t;
typedef bf16_t v2bf  __attribute__((ext_vector_type(2)));
typedef bf16_t v8bf  __attribute__((ext_vector_type(8)));
typedef bf16_t v16bf __attribute__((ext_vector_type(16)));
typedef float  v2f   __attribute__((ext_vector_type(2)));
typedef float  v8f   __attribute__((ext_vector_type(8)));

// native f32 -> bf16 (RNE), lets the backend use its bf16 cvt ops
__device__ __forceinline__ unsigned short f2bf(float f) {
  return __builtin_bit_cast(unsigned short, (bf16_t)f);
}
// packed pair: should lower to a single packed cvt on gfx1250
__device__ __forceinline__ unsigned pack2bf(float x, float y) {
  v2f f = {x, y};
  v2bf b = __builtin_convertvector(f, v2bf);
  return __builtin_bit_cast(unsigned, b);
}
__device__ __forceinline__ v16bf cat8(v8bf l, v8bf h) {
  return __builtin_shufflevector(l, h, 0,1,2,3,4,5,6,7,8,9,10,11,12,13,14,15);
}
// A fragment (16x32 bf16, MxK): per ISA 7.12.2 — lane lo half reads K = c..c+7 and c+16..c+23
__device__ __forceinline__ v16bf load_a_frag(const unsigned short* p, int stride,
                                             int mbase, int ks, int lo, int hi) {
  const unsigned short* r = p + (mbase + lo) * stride + hi * 8 + ks * 32;
  v8bf l = *(const v8bf*)(r);
  v8bf h = *(const v8bf*)(r + 16);
  return cat8(l, h);
}
// B fragment (32x16 bf16, KxN) read from B^T stored row-major [N x K]:
// lane half hi holds K = 8*hi..8*hi+7 and 16+8*hi..16+8*hi+7 (same order as A)
__device__ __forceinline__ v16bf load_b_frag(const unsigned short* p, int stride,
                                             int nbase, int ks, int lo, int hi) {
  const unsigned short* r = p + (nbase + lo) * stride + hi * 8 + ks * 32;
  v8bf l = *(const v8bf*)(r);
  v8bf h = *(const v8bf*)(r + 16);
  return cat8(l, h);
}
__device__ __forceinline__ v8f wmma_bf16(v16bf a, v16bf b, v8f c) {
  return __builtin_amdgcn_wmma_f32_16x16x32_bf16(false, a, false, b, (short)0, c, false, false);
}
__device__ __forceinline__ void lds_fence() { asm volatile("" ::: "memory"); }

// ---------- prologue: f32 weights -> bf16, B^T row-major layouts ----------
// ws layout (ushort): [0]=WqT[4][64][64], [16384]=WkT, [32768]=WvT, [49152]=WresT[256][64]
__global__ __launch_bounds__(256)
void convert_weights_kernel(const float* __restrict__ Wq, const float* __restrict__ Wk,
                            const float* __restrict__ Wv, const float* __restrict__ Wres,
                            unsigned short* __restrict__ ws) {
  int i = blockIdx.x * 256 + threadIdx.x;    // 0..65535
  int seg = i >> 14;
  int j = i & 16383;
  if (seg < 3) {
    const float* W = (seg == 0) ? Wq : (seg == 1) ? Wk : Wv;
    int h = j >> 12, a = (j >> 6) & 63, e = j & 63;
    ws[seg * 16384 + (h * 64 + a) * 64 + e] = f2bf(W[(h * 64 + e) * 64 + a]);
  } else {
    int c = j >> 6, e = j & 63;
    ws[49152 + c * 64 + e] = f2bf(Wres[e * 256 + c]);
  }
}

// ---------------- main fused kernel: one block per batch element ----------------
__global__ __launch_bounds__(128)
void autoint_mha_kernel(const float* __restrict__ X,
                        const unsigned short* __restrict__ wqT,
                        const unsigned short* __restrict__ wkT,
                        const unsigned short* __restrict__ wvT,
                        const unsigned short* __restrict__ wresT,
                        float* __restrict__ out) {
  __shared__ __align__(16) unsigned short smem[LDS_USH];
  const int b    = blockIdx.x;
  const int tid  = threadIdx.x;
  const int lane = tid & 31;
  const int lo   = lane & 15;
  const int hi   = lane >> 4;
  const int h    = tid >> 5;                 // wave id == head id

  // ---- init: zero attn+VT block, zero Xs pad rows, stage X (f32 -> bf16) ----
  {
    uint4 z4 = make_uint4(0u, 0u, 0u, 0u);
    uint4* zp = (uint4*)(smem + ZB_OFF);
    for (int i = tid; i < 4032; i += 128) zp[i] = z4;        // attn[4]+VT[4]
    uint4* xz = (uint4*)(smem + 40 * STR);
    for (int i = tid; i < 72; i += 128) xz[i] = z4;          // Xs rows 40..47
    const float4* xg = (const float4*)(X + (size_t)b * F_DIM * E_DIM);
    for (int i = tid; i < 640; i += 128) {
      float4 v = xg[i];
      int f = i >> 4;                // 16 float4 per 64-wide row
      int e4 = (i & 15) << 2;
      uint2 pk = make_uint2(pack2bf(v.x, v.y), pack2bf(v.z, v.w));
      *(uint2*)(smem + f * STR + e4) = pk;
    }
  }
  __syncthreads();

  // ---- phase 1: Q = X*Wq_h, K = X*Wk_h (row-major in LDS), V = X*Wv_h (transposed) ----
  {
    v16bf ax[3][2];
#pragma unroll
    for (int mt = 0; mt < 3; ++mt)
#pragma unroll
      for (int ks = 0; ks < 2; ++ks)
        ax[mt][ks] = load_a_frag(smem + XS_OFF, STR, mt * 16, ks, lo, hi);

    const unsigned short* WT[3] = { wqT + h * 4096, wkT + h * 4096, wvT + h * 4096 };
#pragma unroll
    for (int mat = 0; mat < 3; ++mat) {
#pragma unroll
      for (int nt = 0; nt < 4; ++nt) {
        v16bf b0 = load_b_frag(WT[mat], 64, nt * 16, 0, lo, hi);
        v16bf b1 = load_b_frag(WT[mat], 64, nt * 16, 1, lo, hi);
#pragma unroll
        for (int mt = 0; mt < 3; ++mt) {
          v8f acc = {};
          acc = wmma_bf16(ax[mt][0], b0, acc);
          acc = wmma_bf16(ax[mt][1], b1, acc);
          if (mat < 2) {
            unsigned short* dst = smem + ((mat == 0) ? QS_OFF(h) : KS_OFF(h));
#pragma unroll
            for (int r = 0; r < 8; ++r)
              dst[(mt * 16 + 8 * hi + r) * STR + nt * 16 + lo] = f2bf(acc[r]);
          } else {                     // V: store transposed VT[a][f], packed pairs
            int a  = nt * 16 + lo;
            int fb = mt * 16 + 8 * hi;
#pragma unroll
            for (int r = 0; r < 8; r += 2) {
              *(unsigned*)(smem + VT_OFF(h) + a * STR + fb + r) =
                  pack2bf(acc[r], acc[r + 1]);
            }
          }
        }
      }
    }
  }
  lds_fence();

  // ---- phase 2: scores = Q * K^T  (preload all frags, then overwrite Q/K region with f32) ----
  float* Sf = (float*)(smem + QS_OFF(h));    // 48 x SSTR f32, aliases dead Qs/Ks
  {
    v16bf aq[3][2], bk[3][2];
#pragma unroll
    for (int t = 0; t < 3; ++t)
#pragma unroll
      for (int ks = 0; ks < 2; ++ks) {
        aq[t][ks] = load_a_frag(smem + QS_OFF(h), STR, t * 16, ks, lo, hi);
        bk[t][ks] = load_b_frag(smem + KS_OFF(h), STR, t * 16, ks, lo, hi);
      }
    lds_fence();
#pragma unroll
    for (int mt = 0; mt < 3; ++mt)
#pragma unroll
      for (int nt = 0; nt < 3; ++nt) {
        v8f acc = {};
        acc = wmma_bf16(aq[mt][0], bk[nt][0], acc);
        acc = wmma_bf16(aq[mt][1], bk[nt][1], acc);
#pragma unroll
        for (int r = 0; r < 8; ++r)
          Sf[(mt * 16 + 8 * hi + r) * SSTR + nt * 16 + lo] = acc[r];
      }
  }
  lds_fence();

  // ---- phase 3: row softmax in f32, emit bf16 attn (cols>=40 stay zero) ----
  for (int it = 0; it < 2; ++it) {
    int row = lane + 32 * it;
    if (row < F_DIM) {
      float* s = Sf + row * SSTR;
      float m = -1e30f;
      for (int c = 0; c < F_DIM; ++c) m = fmaxf(m, s[c]);
      float sum = 0.f;
      for (int c = 0; c < F_DIM; ++c) { float e = __expf(s[c] - m); s[c] = e; sum += e; }
      float inv = 1.f / sum;
      for (int c = 0; c < F_DIM; c += 2)
        *(unsigned*)(smem + ATT_OFF(h) + row * STR + c) =
            pack2bf(s[c] * inv, s[c + 1] * inv);
    }
  }
  lds_fence();

  // ---- phase 4: out = relu(attn*V + X*Wres_slice), residual fused into same accumulators ----
  {
    const unsigned short* wres_h = wresT + (h * 64) * 64;   // rows c = h*64 ..
#pragma unroll
    for (int nt = 0; nt < 4; ++nt) {
      v16bf br0 = load_b_frag(wres_h, 64, nt * 16, 0, lo, hi);
      v16bf br1 = load_b_frag(wres_h, 64, nt * 16, 1, lo, hi);
      v16bf bv0 = load_b_frag(smem + VT_OFF(h), STR, nt * 16, 0, lo, hi);
      v16bf bv1 = load_b_frag(smem + VT_OFF(h), STR, nt * 16, 1, lo, hi);
#pragma unroll
      for (int mt = 0; mt < 3; ++mt) {
        v16bf ax0 = load_a_frag(smem + XS_OFF, STR, mt * 16, 0, lo, hi);
        v16bf ax1 = load_a_frag(smem + XS_OFF, STR, mt * 16, 1, lo, hi);
        v16bf aa0 = load_a_frag(smem + ATT_OFF(h), STR, mt * 16, 0, lo, hi);
        v16bf aa1 = load_a_frag(smem + ATT_OFF(h), STR, mt * 16, 1, lo, hi);
        v8f acc = {};
        acc = wmma_bf16(ax0, br0, acc);
        acc = wmma_bf16(ax1, br1, acc);
        acc = wmma_bf16(aa0, bv0, acc);
        acc = wmma_bf16(aa1, bv1, acc);
        int col = h * 64 + nt * 16 + lo;
        float* op = out + (size_t)b * F_DIM * 256 + col;
#pragma unroll
        for (int r = 0; r < 8; ++r) {
          int f = mt * 16 + 8 * hi + r;
          if (f < F_DIM) op[f * 256] = fmaxf(acc[r], 0.f);
        }
      }
    }
  }
}

extern "C" void kernel_launch(void* const* d_in, const int* in_sizes, int n_in,
                              void* d_out, int out_size, void* d_ws, size_t ws_size,
                              hipStream_t stream) {
  const float* X    = (const float*)d_in[0];
  const float* Wq   = (const float*)d_in[1];
  const float* Wk   = (const float*)d_in[2];
  const float* Wv   = (const float*)d_in[3];
  const float* Wres = (const float*)d_in[4];
  unsigned short* ws = (unsigned short*)d_ws;   // needs 131072 bytes

  convert_weights_kernel<<<256, 256, 0, stream>>>(Wq, Wk, Wv, Wres, ws);
  autoint_mha_kernel<<<4096, 128, 0, stream>>>(
      X, ws, ws + 16384, ws + 32768, ws + 49152, (float*)d_out);
}
